// CapsuleNet_52553219833866
// MI455X (gfx1250) — matrix operand order repaired and checked by reading, not compiled, not run on hardware
//
#include <hip/hip_runtime.h>
#include <hip/hip_bf16.h>
#include <stddef.h>
#include <stdint.h>

typedef __attribute__((ext_vector_type(16))) _Float16 v16h;
typedef __attribute__((ext_vector_type(8)))  _Float16 v8h;
typedef __attribute__((ext_vector_type(8)))  float    v8f;
typedef __attribute__((ext_vector_type(4)))  unsigned int v4u;
typedef __attribute__((ext_vector_type(8)))  int      v8i;
typedef __attribute__((ext_vector_type(4)))  int      v4i;

// ---------------------------------------------------------------------------
// Sizes (fixed by the reference)
// ---------------------------------------------------------------------------
#define BATCH 256
#define C1    256           // conv1 out channels == conv2 in channels
#define C2    256           // conv2 out channels
#define TAPS  81            // 9x9
#define KSPLIT 3            // split-K over taps: 3 slices of 27 taps
#define OPIX  36            // 6x6 conv2 output pixels
#define NCAP  1152          // 32*36 input capsules
#define NDIG  10
#define DCAP  8
#define DOUT  16
#define USLICE ((size_t)BATCH * NCAP * DCAP)             // 2,359,296 floats

#define BPANEL_HALVES (32 * 256)                         // one tap, 32 oc, 256 c
#define BPANEL_BYTES  (BPANEL_HALVES * 2)                // 16384

// workspace layout (bytes, all 256-aligned)
#define WT_BYTES   ((size_t)TAPS * C2 * C1 * 2)          // 10,616,832
#define FEA_BYTES  ((size_t)400 * BATCH * C1 * 2)        // 52,428,800
#define WT_OFF     ((size_t)0)
#define FEA_OFF    (WT_OFF + WT_BYTES)
#define PART_OFF   (FEA_OFF + FEA_BYTES)                 // 3 slices; slice 0 becomes u_raw

// ---------------------------------------------------------------------------
// Kernel 0: transpose conv2 weights  [oc][c][tap] f32 -> wT[tap][oc][c] f16
// ---------------------------------------------------------------------------
__global__ __launch_bounds__(256) void prep_wt_kernel(
    const float* __restrict__ w2, _Float16* __restrict__ wT) {
  int t = blockIdx.x * 256 + threadIdx.x;        // t = tap*65536 + oc*256 + c
  int c   = t & 255;
  int oc  = (t >> 8) & 255;
  int tap = t >> 16;                             // 0..80
  wT[t] = (_Float16)w2[((size_t)oc * C1 + c) * TAPS + tap];
}

// ---------------------------------------------------------------------------
// Kernel 1: conv1 (1->256ch, 9x9, valid) + bias + ReLU, store f16 [y][x][b][c]
// ---------------------------------------------------------------------------
__global__ __launch_bounds__(256) void conv1_kernel(
    const float* __restrict__ img, const float* __restrict__ w1,
    const float* __restrict__ b1, _Float16* __restrict__ fea) {
  int c = threadIdx.x;                 // output channel
  int b = blockIdx.x & 255;            // batch
  int p = blockIdx.x >> 8;             // pixel 0..399
  int y = p / 20, x = p % 20;
  const float* ip = img + (size_t)b * 784;   // 28*28, single input channel
  const float* wp = w1 + c * TAPS;
  float acc = b1[c];
#pragma unroll
  for (int ky = 0; ky < 9; ++ky)
#pragma unroll
    for (int kx = 0; kx < 9; ++kx)
      acc = fmaf(ip[(y + ky) * 28 + x + kx], wp[ky * 9 + kx], acc);
  acc = fmaxf(acc, 0.0f);
  fea[((size_t)p * BATCH + b) * C1 + c] = (_Float16)acc;
}

// ---------------------------------------------------------------------------
// TDM descriptor: 1-D copy of one 16KB B panel (wT[tap][oc0..+31][*]) to LDS.
// D# packing per CDNA5 ISA sec 8.3/8.4 (count=1, type=2, data_size=2B,
// tensor_dim0 = tile_dim0 = 8192 elements, dim1 = 1).
// ---------------------------------------------------------------------------
__device__ __forceinline__ void tdm_load_bpanel(const _Float16* gsrc,
                                                unsigned lds_off) {
  uint64_t ga = (uint64_t)(uintptr_t)gsrc;
  v4u g0;
  g0[0] = 1u;                                        // count=1, user mode
  g0[1] = lds_off;                                   // lds_addr (bytes)
  g0[2] = (unsigned)(ga & 0xffffffffu);              // global_addr[31:0]
  g0[3] = (unsigned)((ga >> 32) & 0x01ffffffu)       // global_addr[56:32]
          | (2u << 30);                              // type=2 ("image")
  v8i g1;
  g1[0] = (int)(1u << 16);                           // data_size=1 (2 bytes)
  g1[1] = (int)(8192u << 16);                        // tensor_dim0[15:0]=8192
  g1[2] = (int)(1u << 16);                           // tensor_dim0 hi=0, tensor_dim1=1
  g1[3] = (int)(8192u << 16);                        // tile_dim0=8192
  g1[4] = 1;                                         // tile_dim1=1, tile_dim2=0
  g1[5] = 8192;                                      // tensor_dim0_stride lo
  g1[6] = 0;                                         // stride0 hi, stride1 lo
  g1[7] = 0;                                         // stride1 hi
  v4i gz4 = {0, 0, 0, 0};
  v8i gz8 = {0, 0, 0, 0, 0, 0, 0, 0};
  __builtin_amdgcn_tensor_load_to_lds(g0, g1, gz4, gz4, gz8, 0);
}

// ---------------------------------------------------------------------------
// Kernel 2: conv2 as implicit GEMM via V_WMMA_F32_16X16X32_F16.
// Block = one (octile, pixel, K-slice); 8 waves = 8 batch tiles of 32.
// B panel (16KB per tap) staged into double-buffered LDS by the Tensor Data
// Mover (async, TENSORcnt), one tap ahead of compute. A fragments stream from
// global. Per wave: 27 taps x 8 k-steps x 4 WMMAs (32x32 register tile).
// ---------------------------------------------------------------------------
__global__ __launch_bounds__(256) void conv2_wmma_kernel(
    const _Float16* __restrict__ fea, const _Float16* __restrict__ wT,
    float* __restrict__ part) {
  __shared__ _Float16 Bsh[2][BPANEL_HALVES];           // 32KB, offset 0 in LDS

  const int wave = threadIdx.x >> 5;   // btile 0..7
  const int lane = threadIdx.x & 31;
  const int h    = lane >> 4;          // half-wave select
  const int col  = lane & 15;

  int octile = blockIdx.x & 7;         // 8 tiles of 32 oc
  int rest   = blockIdx.x >> 3;        // 0..107
  int pix    = rest % OPIX;
  int ks     = rest / OPIX;            // K-slice 0..2
  int oy = pix / 6, ox = pix % 6;
  int b0 = wave * 32, oc0 = octile * 32;

  v8f a00 = {}, a01 = {}, a10 = {}, a11 = {};          // acc[Mhalf][Nhalf]

  const int tap_lo = ks * 27, tap_hi = tap_lo + 27;
  const _Float16* wpan = wT + (size_t)oc0 * C1;        // + tap*C2*C1 per tap

  if (wave == 0)                                       // preload first tap
    tdm_load_bpanel(wpan + (size_t)tap_lo * C2 * C1, 0u);

#pragma unroll 1
  for (int tap = tap_lo; tap < tap_hi; ++tap) {
    int cur = (tap - tap_lo) & 1;
    if (wave == 0) {
      if (tap + 1 < tap_hi) {                          // prefetch next tap
        tdm_load_bpanel(wpan + (size_t)(tap + 1) * C2 * C1,
                        (unsigned)((cur ^ 1) * BPANEL_BYTES));
        __builtin_amdgcn_s_wait_tensorcnt(1);          // wait for panel[cur]
      } else {
        __builtin_amdgcn_s_wait_tensorcnt(0);          // last tap: drain all
      }
    }
    __syncthreads();                                   // panel[cur] ready
    // The TDM writes LDS behind the compiler's back (descriptor carries the
    // LDS address as an integer). Tell the optimizer LDS changed so the
    // B-fragment ds_loads are re-issued every tap instead of being CSE'd.
    asm volatile("" : : "s"(&Bsh[0][0]) : "memory");

    int ky = tap / 9, kx = tap % 9;
    int y = 2 * oy + ky, x = 2 * ox + kx;
    // A rows (M = batch): two 16-row panels from global
    const _Float16* A0 = fea + ((size_t)(y * 20 + x) * BATCH + b0 + col) * C1;
    const _Float16* A1 = A0 + 16 * C1;
    // B cols (N = oc): two 16-col panels from LDS
    const _Float16* Bl0 = &Bsh[cur][(size_t)col * C1];
    const _Float16* Bl1 = Bl0 + 16 * C1;
#pragma unroll 2
    for (int c0 = 0; c0 < C1; c0 += 32) {
      union { v16h v; v8h p[2]; } fa0, fa1;
      fa0.p[0] = *(const v8h*)(A0 + c0 + 16 * h);      // K = c0+16h..+7
      fa0.p[1] = *(const v8h*)(A0 + c0 + 16 * h + 8);  // K = c0+16h+8..+15
      fa1.p[0] = *(const v8h*)(A1 + c0 + 16 * h);
      fa1.p[1] = *(const v8h*)(A1 + c0 + 16 * h + 8);
      v16h fb0 = *(const v16h*)(Bl0 + c0 + 16 * h);    // K = c0+16h..+15
      v16h fb1 = *(const v16h*)(Bl1 + c0 + 16 * h);
      a00 = __builtin_amdgcn_wmma_f32_16x16x32_f16(false, fa0.v, false, fb0, (short)0, a00, false, false);
      a01 = __builtin_amdgcn_wmma_f32_16x16x32_f16(false, fa0.v, false, fb1, (short)0, a01, false, false);
      a10 = __builtin_amdgcn_wmma_f32_16x16x32_f16(false, fa1.v, false, fb0, (short)0, a10, false, false);
      a11 = __builtin_amdgcn_wmma_f32_16x16x32_f16(false, fa1.v, false, fb1, (short)0, a11, false, false);
    }
    __syncthreads();                                   // done reading panel[cur]
  }

  // scatter D tiles into capsule layout: part[ks][b][ (oc&31)*36+pix ][ oc>>5 ]
  float* ps = part + (size_t)ks * USLICE;
#pragma unroll
  for (int nh = 0; nh < 2; ++nh) {
    int oc = oc0 + 16 * nh + col;
    int i  = (oc & 31) * OPIX + pix;
    int g  = oc >> 5;
    const v8f* am  = nh ? &a01 : &a00;                 // Mhalf 0
    const v8f* am1 = nh ? &a11 : &a10;                 // Mhalf 1
#pragma unroll
    for (int r = 0; r < 8; ++r) {
      int bA = b0 + r + 8 * h;                         // C/D layout: M = r + 8*half
      int bB = bA + 16;
      ps[((size_t)bA * NCAP + i) * DCAP + g] = (*am)[r];
      ps[((size_t)bB * NCAP + i) * DCAP + g] = (*am1)[r];
    }
  }
}

// ---------------------------------------------------------------------------
// Kernel 2b: deterministic split-K reduction (+conv2 bias), in-place to slice 0
// ---------------------------------------------------------------------------
__global__ __launch_bounds__(256) void reduce_kernel(
    float* __restrict__ part, const float* __restrict__ b2) {
  size_t t = (size_t)blockIdx.x * 256 + threadIdx.x;   // over USLICE
  int d = (int)(t & 7);
  int i = (int)((t >> 3) % NCAP);
  int oc = (d << 5) | (i / OPIX);
  part[t] = part[t] + part[t + USLICE] + part[t + 2 * USLICE] + b2[oc];
}

// ---------------------------------------------------------------------------
// Kernel 3: squash(u) + u_hat + 3-iteration dynamic routing.
// One block of 288 threads per (j,b) pair; each thread owns 4 capsules.
// ---------------------------------------------------------------------------
__global__ __launch_bounds__(288) void routing_kernel(
    const float* __restrict__ u_raw, const float* __restrict__ third,
    float* __restrict__ out) {
  const int j = blockIdx.x % NDIG;
  const int b = blockIdx.x / NDIG;
  const int tid = threadIdx.x;
  const int lane = tid & 31;

  __shared__ float red[17];            // [0..15]=s vector, [16]=softmax denom

  float uh[4][16];                     // u_hat for this thread's 4 capsules
  float blog[4];                       // routing logits

#pragma unroll
  for (int r = 0; r < 4; ++r) {
    int i = tid + 288 * r;             // 288*4 == 1152 exactly
    const float* up = u_raw + ((size_t)b * NCAP + i) * DCAP;
    float uv[DCAP];
    float sn = 0.0f;
#pragma unroll
    for (int d = 0; d < DCAP; ++d) { uv[d] = up[d]; sn += uv[d] * uv[d]; }
    float fac = sn / (1.0f + sn) * __frsqrt_rn(sn);       // squash factor
    const float* tp = third + ((size_t)j * NCAP + i) * (DCAP * DOUT);
#pragma unroll
    for (int e = 0; e < DOUT; ++e) {
      float s = 0.0f;
#pragma unroll
      for (int d = 0; d < DCAP; ++d) s = fmaf(uv[d] * fac, tp[d * DOUT + e], s);
      uh[r][e] = s;
    }
    blog[r] = 0.0f;
  }

  float vfin[16];
  float facS = 0.0f, Csum = 1.0f;
  for (int it = 0; it < 3; ++it) {
    if (tid < 17) red[tid] = 0.0f;
    __syncthreads();

    // local softmax numerators and weighted partial sums
    float ex[4], psum = 0.0f, ps[16];
#pragma unroll
    for (int e = 0; e < DOUT; ++e) ps[e] = 0.0f;
#pragma unroll
    for (int r = 0; r < 4; ++r) {
      ex[r] = __expf(blog[r]);
      psum += ex[r];
#pragma unroll
      for (int e = 0; e < DOUT; ++e) ps[e] = fmaf(ex[r], uh[r][e], ps[e]);
    }
    // wave reduce (wave32), then one LDS atomic per wave
#pragma unroll
    for (int off = 16; off > 0; off >>= 1) {
      psum += __shfl_xor(psum, off, 32);
#pragma unroll
      for (int e = 0; e < DOUT; ++e) ps[e] += __shfl_xor(ps[e], off, 32);
    }
    if (lane == 0) {
      atomicAdd(&red[16], psum);
#pragma unroll
      for (int e = 0; e < DOUT; ++e) atomicAdd(&red[e], ps[e]);
    }
    __syncthreads();

    // v = squash( s / Csum )  (redundant in every thread)
    Csum = red[16];
    float inv = 1.0f / Csum;
    float sn2 = 0.0f;
#pragma unroll
    for (int e = 0; e < DOUT; ++e) {
      float sv = red[e] * inv;
      vfin[e] = sv;
      sn2 = fmaf(sv, sv, sn2);
    }
    facS = sn2 / (1.0f + sn2) * __frsqrt_rn(sn2);
#pragma unroll
    for (int e = 0; e < DOUT; ++e) vfin[e] *= facS;

    if (it < 2) {
#pragma unroll
      for (int r = 0; r < 4; ++r) {
        float dot = 0.0f;
#pragma unroll
        for (int e = 0; e < DOUT; ++e) dot = fmaf(uh[r][e], vfin[e], dot);
        blog[r] += dot;
      }
    }
    __syncthreads();                   // protect red[] before next zeroing
  }

  if (tid < 16)                        // out[b][j][e]
    out[((size_t)b * NDIG + j) * DOUT + tid] = red[tid] / Csum * facS;
}

// ---------------------------------------------------------------------------
extern "C" void kernel_launch(void* const* d_in, const int* in_sizes, int n_in,
                              void* d_out, int out_size, void* d_ws, size_t ws_size,
                              hipStream_t stream) {
  const float* images  = (const float*)d_in[0];   // [256,1,28,28]
  const float* conv1_w = (const float*)d_in[1];   // [256,1,9,9]
  const float* conv1_b = (const float*)d_in[2];   // [256]
  const float* conv2_w = (const float*)d_in[3];   // [256,256,9,9]
  const float* conv2_b = (const float*)d_in[4];   // [256]
  const float* third   = (const float*)d_in[5];   // [10,1152,8,16]
  float* out = (float*)d_out;                     // [256,10,16]

  char* ws = (char*)d_ws;
  _Float16* wT    = (_Float16*)(ws + WT_OFF);
  _Float16* fea16 = (_Float16*)(ws + FEA_OFF);
  float*    part  = (float*)(ws + PART_OFF);      // 3 x USLICE; slice 0 -> u_raw

  // 0) weight transpose to f16 [tap][oc][c]
  prep_wt_kernel<<<(TAPS * C2 * C1) / 256, 256, 0, stream>>>(conv2_w, wT);
  // 1) conv1 + bias + relu -> f16 [y][x][b][c]
  conv1_kernel<<<400 * BATCH, 256, 0, stream>>>(images, conv1_w, conv1_b, fea16);
  // 2) conv2 via WMMA, TDM-staged B panels in LDS, split-K over 3 tap slices
  conv2_wmma_kernel<<<8 * OPIX * KSPLIT, 256, 0, stream>>>(fea16, wT, part);
  // 2b) deterministic split-K reduce (+bias) -> u_raw == part slice 0
  reduce_kernel<<<(int)(USLICE / 256), 256, 0, stream>>>(part, conv2_b);
  // 3) squash + u_hat + dynamic routing -> out[b][j][e]
  routing_kernel<<<BATCH * NDIG, 288, 0, stream>>>(part, third, out);
}
